// StructuredAttention_21947282883099
// MI455X (gfx1250) — matrix operand order repaired and checked by reading, not compiled, not run on hardware
//
#include <hip/hip_runtime.h>
#include <hip/hip_bf16.h>

// MI455X / gfx1250, wave32. Structured attention with the batched 512x512
// inversion + O(N^2*SEM) einsum collapsed analytically (L = diag(a)(N I-11^T)
// with row 0 replaced -> Linv = rank-1 + diagonal). Remaining work: three
// bf16 WMMA GEMMs (~17 GFLOP) vs ~100 MB HBM traffic -> memory-bound at
// 23.3 TB/s (~4.3 us floor). Weights pre-transposed once to bf16 col-major;
// kernel A streams weight panels global->LDS with the CDNA5 async-copy path
// (GLOBAL_LOAD_ASYNC_TO_LDS_B128 + s_wait_asynccnt), double-buffered so the
// DMA overlaps the WMMA pipe; kernel C serves B-fragments straight from L2.
// K-loops fully unrolled so accumulators stay home and WMMAs run in place.

#define SEMN 256
#define STRN 256
#define BBATCH 64
#define NROW 512

typedef __bf16 bf16;
typedef __bf16 v16bf __attribute__((ext_vector_type(16)));
typedef __bf16 v8bf  __attribute__((ext_vector_type(8)));
typedef __bf16 v4bf  __attribute__((ext_vector_type(4)));
typedef float  v8f   __attribute__((ext_vector_type(8)));
typedef float  v4f   __attribute__((ext_vector_type(4)));

__device__ __forceinline__ v16bf cat8(v8bf lo, v8bf hi) {
  return __builtin_shufflevector(lo, hi, 0, 1, 2, 3, 4, 5, 6, 7, 8, 9, 10, 11,
                                 12, 13, 14, 15);
}

// ---- CDNA5 async global->LDS copy (ASYNCcnt) ------------------------------
__device__ __forceinline__ uint32_t lds_addr32(const void* p) {
  return (uint32_t)(uintptr_t)p;  // generic shared ptr: low 32 bits = LDS addr
}
__device__ __forceinline__ void async_b128(uint32_t lds_off, const void* g) {
  asm volatile("global_load_async_to_lds_b128 %0, %1, off" ::"v"(lds_off),
               "v"(g)
               : "memory");
}
__device__ __forceinline__ void wait_async0() {
  asm volatile("s_wait_asynccnt 0x0" ::: "memory");
}

// A-matrix 16x32 bf16 fragment (ISA 7.12.2): lane L = row m=L%16, half
// h=L/16; element e<8 -> K=h*8+e, e>=8 -> K=16+h*8+(e-8). Two aligned 16B
// chunks of a row-major tile -> 2x ds_load_b128.
__device__ __forceinline__ v16bf frag_a(const bf16* tile, int lda, int lane) {
  const int m = lane & 15, h = lane >> 4;
  const bf16* p = tile + m * lda + h * 8;
  const v8bf lo = *(const v8bf*)p;
  const v8bf hi = *(const v8bf*)(p + 16);
  return cat8(lo, hi);
}

// B-matrix 32x16 bf16 fragment from an LDS panel pan[256 cols][32 k]:
// lane L = col n=L%16, e<8 -> K=h*8+e, e>=8 -> K=16+h*8+(e-8) -> 2x ds_load_b128.
__device__ __forceinline__ v16bf frag_b_p(const bf16* pan, int col, int lane) {
  const int n = lane & 15, h = lane >> 4;
  const bf16* p = pan + (col + n) * 32 + h * 8;
  const v8bf lo = *(const v8bf*)p;
  const v8bf hi = *(const v8bf*)(p + 16);
  return cat8(lo, hi);
}

// B fragment straight from bf16 column-major weights Wt[col][K] in global.
__device__ __forceinline__ v16bf frag_b_t(const bf16* __restrict__ Wt, int K,
                                          int col, int k0, int lane) {
  const int n = lane & 15, h = lane >> 4;
  const bf16* p = Wt + (size_t)(col + n) * K + k0 + h * 8;
  const v8bf lo = *(const v8bf*)p;
  const v8bf hi = *(const v8bf*)(p + 16);
  return cat8(lo, hi);
}

// Async-stage one 32x256 K-panel of col-major Wt into pan[256][32] (16 KB):
// 1024 16B chunks, 4 per thread.
__device__ __forceinline__ void stage_panel_async(const bf16* __restrict__ Wt,
                                                  int K, int k0, bf16* pan,
                                                  int tid) {
#pragma unroll
  for (int i = 0; i < 4; ++i) {
    const int q = tid + 256 * i;
    const int c = q >> 2, ch = q & 3;
    async_b128(lds_addr32(pan + c * 32 + ch * 8),
               Wt + (size_t)c * K + k0 + ch * 8);
  }
}

__device__ __forceinline__ void store4_bf16(bf16* dst, v4f v) {
  v4bf p;
  p[0] = (bf16)v[0]; p[1] = (bf16)v[1]; p[2] = (bf16)v[2]; p[3] = (bf16)v[3];
  *(v4bf*)dst = p;  // 8-byte ds_store_b64
}

// ---------------------------------------------------------------------------
__global__ __launch_bounds__(256) void sa_init(float* G_ws) {
  if (threadIdx.x < BBATCH) G_ws[threadIdx.x] = 0.f;
}

// One-shot: bf16 column-major copies of the three weight matrices.
__global__ __launch_bounds__(256) void sa_kW(
    const float* __restrict__ tp_w, const float* __restrict__ bl_w,
    const float* __restrict__ fz_w, bf16* __restrict__ tpwt,
    bf16* __restrict__ blwt, bf16* __restrict__ fzwt) {
  const int idx = blockIdx.x * 256 + threadIdx.x;  // grid covers 131072
  if (idx < 65536) {
    const int c = idx >> 8, k = idx & 255;
    tpwt[idx] = (bf16)tp_w[k * 256 + c];
    blwt[idx] = (bf16)bl_w[k * 256 + c];
  }
  if (idx < 131072) {
    const int c = idx >> 9, k = idx & 511;
    fzwt[idx] = (bf16)fz_w[k * 256 + c];
  }
}

// Kernel A: per (b, 64-row tile): tp = str@tp_w + tp_b (WMMA), q = tp@bl_w
// (WMMA), f_row = sum(tp*q), a = exp(f_row+bl_b), g = exp(str.fi_w+fi_b),
// G[b] += sum(g). Weight panels double-buffered via async global->LDS DMA.
__global__ __launch_bounds__(256) void sa_kA(
    const float* __restrict__ x, const bf16* __restrict__ tpwt,
    const float* __restrict__ tp_b, const bf16* __restrict__ blwt,
    const float* __restrict__ bl_b, const float* __restrict__ fi_w,
    const float* __restrict__ fi_b, float* __restrict__ a_ws,
    float* __restrict__ g_ws, float* __restrict__ G_ws) {
  extern __shared__ char smem[];
  const int LDA = STRN + 8;                          // 264 elems, 528 B
  bf16* sA   = (bf16*)smem;                          // [64][264] str tile
  bf16* tpB  = (bf16*)(smem + (size_t)64 * LDA * 2); // [64][264] tp tile
  bf16* pan0 = (bf16*)(smem + (size_t)128 * LDA * 2);  // [256][32]
  bf16* pan1 = pan0 + 256 * 32;                        // [256][32]
  float* frow = (float*)(pan1 + 256 * 32);             // [64]

  const int b = blockIdx.x, t = blockIdx.y;
  const int tid = threadIdx.x, lane = tid & 31, w = tid >> 5;
  const int row0 = t * 64;
  const int rb = w >> 1;        // 4 row blocks of 16
  const int cb0 = (w & 1) * 8;  // 2 waves split 16 col blocks

  // kick off DMA of the first tp_w panel, then stage the str tile
  stage_panel_async(tpwt, STRN, 0, pan0, tid);
  for (int e = tid; e < 64 * (STRN / 4); e += 256) {
    const int r = e >> 6, c4 = (e & 63) * 4;
    const v4f v =
        *(const v4f*)&x[((size_t)b * NROW + row0 + r) * (SEMN + STRN) + SEMN + c4];
    store4_bf16(&sA[r * LDA + c4], v);
  }
  if (tid < 64) frow[tid] = 0.f;
  wait_async0();
  __syncthreads();

  // GEMM1: tp = str @ tp_w   (K=256, B panels via async DMA, double-buffered)
  {
    v8f acc[8];
#pragma unroll
    for (int i = 0; i < 8; ++i)
#pragma unroll
      for (int e = 0; e < 8; ++e) acc[i][e] = 0.f;
#pragma unroll
    for (int kp = 0; kp < STRN / 32; ++kp) {
      bf16* cur = (kp & 1) ? pan1 : pan0;
      bf16* nxt = (kp & 1) ? pan0 : pan1;
      if (kp + 1 < STRN / 32)
        stage_panel_async(tpwt, STRN, (kp + 1) * 32, nxt, tid);
      const v16bf af = frag_a(&sA[(rb * 16) * LDA + kp * 32], LDA, lane);
#pragma unroll
      for (int c = 0; c < 8; ++c) {
        const v16bf bfr = frag_b_p(cur, (cb0 + c) * 16, lane);
        acc[c] = __builtin_amdgcn_wmma_f32_16x16x32_bf16(
            false, af, false, bfr, (short)0, acc[c], false, false);
      }
      wait_async0();
      __syncthreads();
    }
    // prefetch first bl_w panel while adding bias + spilling tp (bf16) -> LDS
    stage_panel_async(blwt, STRN, 0, pan0, tid);
    {
      const int h = lane >> 4, n = lane & 15;
#pragma unroll
      for (int c = 0; c < 8; ++c) {
        const int col = (cb0 + c) * 16 + n;
        const float bias = tp_b[col];
#pragma unroll
        for (int r = 0; r < 8; ++r)
          tpB[(rb * 16 + r + 8 * h) * LDA + col] = (bf16)(acc[c][r] + bias);
      }
    }
    wait_async0();
    __syncthreads();
  }  // acc (GEMM1) dies here: only one 64-VGPR accumulator set live at a time

  // GEMM2: q = tp @ bl_w ; then f_row = sum(tp*q) (tp re-read from LDS spill)
  {
    v8f acc[8];
#pragma unroll
    for (int i = 0; i < 8; ++i)
#pragma unroll
      for (int e = 0; e < 8; ++e) acc[i][e] = 0.f;
#pragma unroll
    for (int kp = 0; kp < STRN / 32; ++kp) {
      bf16* cur = (kp & 1) ? pan1 : pan0;
      bf16* nxt = (kp & 1) ? pan0 : pan1;
      if (kp + 1 < STRN / 32)
        stage_panel_async(blwt, STRN, (kp + 1) * 32, nxt, tid);
      const v16bf af = frag_a(&tpB[(rb * 16) * LDA + kp * 32], LDA, lane);
#pragma unroll
      for (int c = 0; c < 8; ++c) {
        const v16bf bfr = frag_b_p(cur, (cb0 + c) * 16, lane);
        acc[c] = __builtin_amdgcn_wmma_f32_16x16x32_bf16(
            false, af, false, bfr, (short)0, acc[c], false, false);
      }
      wait_async0();
      __syncthreads();
    }
    {
      const int h = lane >> 4, n = lane & 15;
      float part[8];
#pragma unroll
      for (int r = 0; r < 8; ++r) part[r] = 0.f;
#pragma unroll
      for (int c = 0; c < 8; ++c) {
        const int col = (cb0 + c) * 16 + n;
#pragma unroll
        for (int r = 0; r < 8; ++r)
          part[r] +=
              acc[c][r] * (float)tpB[(rb * 16 + r + 8 * h) * LDA + col];
      }
#pragma unroll
      for (int r = 0; r < 8; ++r)
        atomicAdd(&frow[rb * 16 + r + 8 * h], part[r]);
    }
  }
  __syncthreads();

  if (tid < 64) {
    const int m = tid;
    const float a = __expf(frow[m] + bl_b[0]);
    a_ws[(size_t)b * NROW + row0 + m] = a;
    float gd = fi_b[0];
    for (int d0 = 0; d0 < STRN; d0 += 8) {
      const v8bf s8 = *(const v8bf*)&sA[m * LDA + d0];
#pragma unroll
      for (int e = 0; e < 8; ++e) gd += (float)s8[e] * fi_w[d0 + e];
    }
    const float g = __expf(gd);
    g_ws[(size_t)b * NROW + row0 + m] = g;
    atomicAdd(&G_ws[b], g);
  }
}

// Kernel B: per-batch rank-1 sums S_u/S_v/S_w over the sem channels.
__global__ __launch_bounds__(256) void sa_kB(
    const float* __restrict__ x, const float* __restrict__ a_ws,
    const float* __restrict__ g_ws, const float* __restrict__ G_ws,
    float* __restrict__ Su, float* __restrict__ Sv, float* __restrict__ Sw) {
  __shared__ float us[NROW], vs[NROW], wx[NROW];
  const int b = blockIdx.x, tid = threadIdx.x;
  const float G = G_ws[b], g0 = g_ws[(size_t)b * NROW];
  for (int i = tid; i < NROW; i += 256) {
    const float a = a_ws[(size_t)b * NROW + i];
    const float g = g_ws[(size_t)b * NROW + i];
    const float na = (float)NROW * a;
    float x0, c, dd, m;
    if (i == 0) { x0 = 1.f / G; c = x0; dd = x0; m = 0.f; }
    else {
      x0 = -(G - g0 + g) / (na * G);
      c  = x0 + 1.f / na;
      dd = c + 1.f / na;
      m  = 1.f;
    }
    us[i] = a * dd;
    vs[i] = a * c * m;
    wx[i] = a * x0 * m;
  }
  __syncthreads();
  const int s = tid;
  float su = 0.f, sv = 0.f, sw = 0.f;
  for (int i = 0; i < NROW; ++i) {
    const float sem = x[((size_t)b * NROW + i) * (SEMN + STRN) + s];
    su += us[i] * sem;
    sv += vs[i] * sem;
    sw += wx[i] * sem;
  }
  Su[b * SEMN + s] = su;
  Sv[b * SEMN + s] = sv;
  Sw[b * SEMN + s] = sw;
}

// Kernel C: pinp in closed form, finp = [sem | pinp], out = relu(finp@fz_w+b)
__global__ __launch_bounds__(256) void sa_kC(
    const float* __restrict__ x, const float* __restrict__ exparam,
    const bf16* __restrict__ fzwt, const float* __restrict__ fz_b,
    const float* __restrict__ a_ws, const float* __restrict__ g_ws,
    const float* __restrict__ G_ws, const float* __restrict__ Su,
    const float* __restrict__ Sv, const float* __restrict__ Sw,
    float* __restrict__ out) {
  extern __shared__ char smem[];
  const int LDF = 512 + 8;  // 1040 B rows, 16B aligned
  bf16* fin = (bf16*)smem;  // [64][520]

  const int b = blockIdx.x, t = blockIdx.y;
  const int tid = threadIdx.x, lane = tid & 31, w = tid >> 5;
  const int row0 = t * 64;
  const float G = G_ws[b], g0 = g_ws[(size_t)b * NROW];
  const float exv = exparam[tid];
  const float su = Su[b * SEMN + tid], sv = Sv[b * SEMN + tid],
              sw = Sw[b * SEMN + tid];

  for (int j = 0; j < 64; ++j) {
    const int n = row0 + j;
    const float a = a_ws[(size_t)b * NROW + n];
    const float g = g_ws[(size_t)b * NROW + n];
    const float semv = x[((size_t)b * NROW + n) * (SEMN + STRN) + tid];
    float pv;
    if (n == 0) {
      pv = (g / G) * exv - sw;
    } else {
      const float na = (float)NROW * a;
      const float x0 = -(G - g0 + g) / (na * G);
      const float c  = x0 + 1.f / na;
      const float dd = c + 1.f / na;
      const float uv = a * (dd - c);  // == 1/N analytically
      pv = (g / G) * exv + su - sv - uv * semv;
    }
    fin[j * LDF + tid]        = (bf16)semv;
    fin[j * LDF + SEMN + tid] = (bf16)pv;
  }
  __syncthreads();

  const int rb = w >> 1, cb0 = (w & 1) * 8;
  v8f acc[8];
#pragma unroll
  for (int i = 0; i < 8; ++i)
#pragma unroll
    for (int e = 0; e < 8; ++e) acc[i][e] = 0.f;

#pragma unroll
  for (int kp = 0; kp < 512 / 32; ++kp) {
    const v16bf af = frag_a(&fin[(rb * 16) * LDF + kp * 32], LDF, lane);
#pragma unroll
    for (int c = 0; c < 8; ++c) {
      const v16bf bfr = frag_b_t(fzwt, 512, (cb0 + c) * 16, kp * 32, lane);
      acc[c] = __builtin_amdgcn_wmma_f32_16x16x32_bf16(
          false, af, false, bfr, (short)0, acc[c], false, false);
    }
  }
  {
    const int h = lane >> 4, nn = lane & 15;
#pragma unroll
    for (int c = 0; c < 8; ++c) {
      const int col = (cb0 + c) * 16 + nn;
      const float bias = fz_b[col];
#pragma unroll
      for (int r = 0; r < 8; ++r) {
        const float v = acc[c][r] + bias;
        out[((size_t)b * NROW + row0 + rb * 16 + r + 8 * h) * SEMN + col] =
            v > 0.f ? v : 0.f;
      }
    }
  }
}

// ---------------------------------------------------------------------------
extern "C" void kernel_launch(void* const* d_in, const int* in_sizes, int n_in,
                              void* d_out, int out_size, void* d_ws,
                              size_t ws_size, hipStream_t stream) {
  const float* x    = (const float*)d_in[0];
  const float* tp_w = (const float*)d_in[1];
  const float* tp_b = (const float*)d_in[2];
  const float* bl_w = (const float*)d_in[3];
  const float* bl_b = (const float*)d_in[4];
  const float* fi_w = (const float*)d_in[5];
  const float* fi_b = (const float*)d_in[6];
  const float* expm = (const float*)d_in[7];
  const float* fz_w = (const float*)d_in[8];
  const float* fz_b = (const float*)d_in[9];
  float* out = (float*)d_out;

  float* ws = (float*)d_ws;
  float* a_ws = ws;                    // B*N
  float* g_ws = a_ws + BBATCH * NROW;  // B*N
  float* G_ws = g_ws + BBATCH * NROW;  // B
  float* Su   = G_ws + BBATCH;         // B*SEM
  float* Sv   = Su + BBATCH * SEMN;
  float* Sw   = Sv + BBATCH * SEMN;
  // bf16 column-major weights after the float region (64B-aligned offset)
  bf16* tpwt = (bf16*)(Sw + BBATCH * SEMN);  // 256*256
  bf16* blwt = tpwt + 256 * 256;             // 256*256
  bf16* fzwt = blwt + 256 * 256;             // 256*512

  const int LDA = STRN + 8, LDF = 512 + 8;
  const size_t shA =
      (size_t)128 * LDA * 2 + 2 * 256 * 32 * 2 + 64 * 4;  // ~100 KB
  const size_t shC = (size_t)64 * LDF * 2;                // ~65 KB

  sa_init<<<1, 256, 0, stream>>>(G_ws);
  sa_kW<<<512, 256, 0, stream>>>(tp_w, bl_w, fz_w, tpwt, blwt, fzwt);
  sa_kA<<<dim3(BBATCH, NROW / 64), 256, shA, stream>>>(
      x, tpwt, tp_b, blwt, bl_b, fi_w, fi_b, a_ws, g_ws, G_ws);
  sa_kB<<<dim3(BBATCH), 256, 0, stream>>>(x, a_ws, g_ws, G_ws, Su, Sv, Sw);
  sa_kC<<<dim3(BBATCH, NROW / 64), 256, shC, stream>>>(
      x, expm, fzwt, fz_b, a_ws, g_ws, G_ws, Su, Sv, Sw, out);
}
